// GPTBlock_40123584479640
// MI455X (gfx1250) — compile-verified, matrix operand order checked by reading, not run on hardware
//
#include <hip/hip_runtime.h>
#include <hip/hip_bf16.h>
#include <math.h>

typedef __attribute__((ext_vector_type(16))) _Float16 v16h;
typedef __attribute__((ext_vector_type(8)))  float    v8f;
typedef __attribute__((ext_vector_type(8)))  _Float16 h8;

enum { FLAG_GELU = 1, FLAG_HEADSPLIT = 2, FLAG_VT = 4 };

__device__ __forceinline__ v8f wmma_f16(v16h a, v16h b, v8f c) {
    return __builtin_amdgcn_wmma_f32_16x16x32_f16(false, a, false, b,
                                                  (short)0, c, false, false);
}

// Build a 16x32 (A) / 32x16 (B) f16 fragment from a row-major 32-element row.
// Element e=0..7  <-> K = 8*hh + e      (contiguous -> b128 load)
// Element e=8..15 <-> K = 16 + 8*hh + e-8
__device__ __forceinline__ v16h frag_row32(const _Float16* rowbase, int hh) {
    v16h f;
    h8 lo = *(const h8*)(rowbase + 8 * hh);
    h8 hi = *(const h8*)(rowbase + 16 + 8 * hh);
#pragma unroll
    for (int i = 0; i < 8; ++i) { f[i] = lo[i]; f[8 + i] = hi[i]; }
    return f;
}

// ---------------------------------------------------------------------------
// f32 [K][N] -> f16 transposed [N][K] (tiled through LDS, coalesced both ways)
// ---------------------------------------------------------------------------
__global__ __launch_bounds__(256) void convert_transpose_f16(
    const float* __restrict__ in, _Float16* __restrict__ out, int K, int N) {
    __shared__ float tile[32][33];
    int k0 = blockIdx.y * 32, n0 = blockIdx.x * 32;
    int tx = threadIdx.x & 31, ty = threadIdx.x >> 5;   // 32 x 8
#pragma unroll
    for (int i = ty; i < 32; i += 8)
        tile[i][tx] = in[(size_t)(k0 + i) * N + n0 + tx];
    __syncthreads();
#pragma unroll
    for (int i = ty; i < 32; i += 8)
        out[(size_t)(n0 + i) * K + k0 + tx] = (_Float16)tile[tx][i];
}

// ---------------------------------------------------------------------------
// LayerNorm over rows of C cols, f32 in, f16 out
// ---------------------------------------------------------------------------
__global__ __launch_bounds__(256) void layernorm_f16(
    const float* __restrict__ x, const float* __restrict__ g,
    const float* __restrict__ bta, _Float16* __restrict__ out, int C) {
    int row = blockIdx.x;
    const float* xr = x + (size_t)row * C;
    float s = 0.f, s2 = 0.f;
    for (int i = threadIdx.x; i < C; i += 256) {
        float v = xr[i];
        s += v; s2 += v * v;
    }
    __shared__ float rs[256], rs2[256];
    rs[threadIdx.x] = s; rs2[threadIdx.x] = s2;
    __syncthreads();
    for (int off = 128; off > 0; off >>= 1) {
        if (threadIdx.x < (unsigned)off) {
            rs[threadIdx.x]  += rs[threadIdx.x + off];
            rs2[threadIdx.x] += rs2[threadIdx.x + off];
        }
        __syncthreads();
    }
    float mean = rs[0] / C;
    float var  = rs2[0] / C - mean * mean;
    float rstd = rsqrtf(var + 1e-5f);
    for (int i = threadIdx.x; i < C; i += 256) {
        float v = (xr[i] - mean) * rstd * g[i] + bta[i];
        out[(size_t)row * C + i] = (_Float16)v;
    }
}

// ---------------------------------------------------------------------------
// WMMA GEMM:  C[M,N] = A[M,K](f16) * BT[N,K]^T(f16)  (+bias, gelu, resid)
// Block: 256 threads = 8 waves (2x4), block tile 64(M) x 128(N), K-step 32.
// Each wave: 32x32 -> 4 WMMAs per K-step with 2x2 fragment reuse.
// All staging and fragment traffic is b128 (BT is pre-transposed [N][K]).
// Software-pipelined: next global tiles fetched during compute.
// ---------------------------------------------------------------------------
__global__ __launch_bounds__(256) void gemm_wmma(
    const _Float16* __restrict__ A, const _Float16* __restrict__ BT,
    const float* __restrict__ bias, const float* __restrict__ resid,
    float* __restrict__ Cf, _Float16* __restrict__ Ch,
    int M, int N, int K, int Tdim, int flags) {
    __shared__ __align__(16) _Float16 As[64][40];    // [m][k], 80B row stride
    __shared__ __align__(16) _Float16 Bs[128][40];   // [n][k]

    int tid  = threadIdx.x;
    int wid  = tid >> 5, lane = tid & 31;
    int hh   = lane >> 4, ln = lane & 15;
    int wm   = wid >> 2, wn = wid & 3;               // 2 x 4 wave grid
    int m0   = blockIdx.y * 64, n0 = blockIdx.x * 128;

    int arow = tid >> 2, acol = (tid & 3) << 3;      // A tile: 64 x 32, h8/thread
    int brow = tid >> 1, bcol = (tid & 1) << 4;      // B tile: 128 x 32, 2 h8/thread

    const _Float16* Aptr = A  + (size_t)(m0 + arow) * K + acol;
    const _Float16* Bptr = BT + (size_t)(n0 + brow) * K + bcol;

    v8f cacc[2][2] = {};

    h8 a_reg  = *(const h8*)(Aptr);
    h8 b_reg0 = *(const h8*)(Bptr);
    h8 b_reg1 = *(const h8*)(Bptr + 8);

    for (int kb = 0; kb < K; kb += 32) {
        *(h8*)&As[arow][acol]     = a_reg;
        *(h8*)&Bs[brow][bcol]     = b_reg0;
        *(h8*)&Bs[brow][bcol + 8] = b_reg1;
        __syncthreads();
        if (kb + 32 < K) {                            // pipeline next tiles
            a_reg  = *(const h8*)(Aptr + kb + 32);
            b_reg0 = *(const h8*)(Bptr + kb + 32);
            b_reg1 = *(const h8*)(Bptr + kb + 40);
        }
        v16h a0 = frag_row32(&As[wm * 32 + ln][0], hh);
        v16h a1 = frag_row32(&As[wm * 32 + 16 + ln][0], hh);
        v16h b0 = frag_row32(&Bs[wn * 32 + ln][0], hh);
        v16h b1 = frag_row32(&Bs[wn * 32 + 16 + ln][0], hh);
        cacc[0][0] = wmma_f16(a0, b0, cacc[0][0]);
        cacc[0][1] = wmma_f16(a0, b1, cacc[0][1]);
        cacc[1][0] = wmma_f16(a1, b0, cacc[1][0]);
        cacc[1][1] = wmma_f16(a1, b1, cacc[1][1]);
        __syncthreads();
    }

#pragma unroll
    for (int ti = 0; ti < 2; ++ti) {
#pragma unroll
        for (int tj = 0; tj < 2; ++tj) {
#pragma unroll
            for (int r = 0; r < 8; ++r) {
                int row = m0 + wm * 32 + ti * 16 + hh * 8 + r;
                int col = n0 + wn * 32 + tj * 16 + ln;
                float v = cacc[ti][tj][r];
                if (bias) v += bias[col];
                if (flags & FLAG_GELU)
                    v = 0.5f * v * (1.0f + erff(v * 0.70710678118f));
                if (resid) v += resid[(size_t)row * N + col];
                if (Cf) Cf[(size_t)row * N + col] = v;
                if (Ch) {
                    if (flags & FLAG_HEADSPLIT) {
                        int t = row % Tdim, bb = row / Tdim;
                        int head = col >> 6, dim = col & 63;
                        int Hh = N >> 6;
                        Ch[(((size_t)(bb * Hh + head) * Tdim) + t) * 64 + dim] =
                            (_Float16)v;
                    } else if (flags & FLAG_VT) {
                        int t = row % Tdim, bb = row / Tdim;
                        int head = col >> 6, dim = col & 63;
                        int Hh = N >> 6;
                        Ch[(((size_t)(bb * Hh + head) * 64) + dim) * Tdim + t] =
                            (_Float16)v;
                    } else {
                        Ch[(size_t)row * N + col] = (_Float16)v;
                    }
                }
            }
        }
    }
}

// ---------------------------------------------------------------------------
// Flash attention, causal, head_dim=64.
// Q,K in [B*H][T][64] f16; V pre-transposed in [B*H][64][T] f16.
// Block: 8 waves; wave owns 32 query rows (2 tiles); 32-key online-softmax
// steps. Per step per wave: 8 score WMMAs + 8 PV WMMAs with K/V frag reuse.
// ---------------------------------------------------------------------------
__global__ __launch_bounds__(256) void attn_wmma(
    const _Float16* __restrict__ Qh, const _Float16* __restrict__ Kh,
    const _Float16* __restrict__ Vt, _Float16* __restrict__ O,
    int T, int H) {
    __shared__ __align__(16) _Float16 Plds[8][2][16][40];  // per-wave P buffers

    int tid  = threadIdx.x;
    int wid  = tid >> 5, lane = tid & 31;
    int hh   = lane >> 4, ln = lane & 15;
    int bh   = blockIdx.y;
    int b    = bh / H, head = bh % H;
    int q0   = blockIdx.x * 256 + wid * 32;
    const float scale = 0.125f;  // 1/sqrt(64)

    // Q fragments: 2 q-tiles x 2 K-dim halves
    v16h aq[2][2];
#pragma unroll
    for (int t = 0; t < 2; ++t) {
        const _Float16* qb = Qh + ((size_t)bh * T + q0 + t * 16 + ln) * 64;
        aq[t][0] = frag_row32(qb, hh);
        aq[t][1] = frag_row32(qb + 32, hh);
    }

    float mrow[2][8], lrow[2][8];
    v8f oacc[2][4] = {};
#pragma unroll
    for (int t = 0; t < 2; ++t)
#pragma unroll
        for (int r = 0; r < 8; ++r) { mrow[t][r] = -1e30f; lrow[t][r] = 0.f; }

    for (int k0 = 0; k0 < q0 + 32; k0 += 32) {
        float sv[2][2][8];
#pragma unroll
        for (int s = 0; s < 2; ++s) {
            int kb = k0 + 16 * s;
            const _Float16* kp = Kh + ((size_t)bh * T + kb + ln) * 64;
            v16h bk0 = frag_row32(kp, hh);
            v16h bk1 = frag_row32(kp + 32, hh);
            int keyg = kb + ln;
#pragma unroll
            for (int t = 0; t < 2; ++t) {
                v8f sc = {};
                sc = wmma_f16(aq[t][0], bk0, sc);
                sc = wmma_f16(aq[t][1], bk1, sc);
#pragma unroll
                for (int r = 0; r < 8; ++r) {
                    int qg = q0 + t * 16 + hh * 8 + r;
                    float vv = sc[r] * scale;
                    sv[t][s][r] = (keyg > qg) ? -1e30f : vv;
                }
            }
        }
        // online softmax per q-tile over this 32-key strip
#pragma unroll
        for (int t = 0; t < 2; ++t) {
#pragma unroll
            for (int r = 0; r < 8; ++r) {
                float rm = fmaxf(sv[t][0][r], sv[t][1][r]);
#pragma unroll
                for (int off = 1; off < 16; off <<= 1)
                    rm = fmaxf(rm, __shfl_xor(rm, off, 16));
                float mnew = fmaxf(mrow[t][r], rm);
                float corr = __expf(mrow[t][r] - mnew);
                float p0 = __expf(sv[t][0][r] - mnew);
                float p1 = __expf(sv[t][1][r] - mnew);
                float rsum = p0 + p1;
#pragma unroll
                for (int off = 1; off < 16; off <<= 1)
                    rsum += __shfl_xor(rsum, off, 16);
                lrow[t][r] = lrow[t][r] * corr + rsum;
                mrow[t][r] = mnew;
#pragma unroll
                for (int d = 0; d < 4; ++d) oacc[t][d][r] *= corr;
                Plds[wid][t][hh * 8 + r][ln]      = (_Float16)p0;
                Plds[wid][t][hh * 8 + r][16 + ln] = (_Float16)p1;
            }
        }
        // wave-local LDS transpose: score rows -> A-fragment layout
        asm volatile("s_wait_dscnt 0" ::: "memory");
        __builtin_amdgcn_wave_barrier();
        v16h ap[2];
#pragma unroll
        for (int t = 0; t < 2; ++t)
            ap[t] = frag_row32(&Plds[wid][t][ln][0], hh);
        __builtin_amdgcn_wave_barrier();
        // P (16x32) x V (32x64): V^T rows are key-contiguous -> b128 loads
#pragma unroll
        for (int d = 0; d < 4; ++d) {
            const _Float16* vp =
                Vt + ((size_t)bh * 64 + d * 16 + ln) * T + k0;
            v16h bv = frag_row32(vp, hh);
            oacc[0][d] = wmma_f16(ap[0], bv, oacc[0][d]);
            oacc[1][d] = wmma_f16(ap[1], bv, oacc[1][d]);
        }
    }

    int C = H * 64;
#pragma unroll
    for (int t = 0; t < 2; ++t)
#pragma unroll
        for (int d = 0; d < 4; ++d)
#pragma unroll
            for (int r = 0; r < 8; ++r) {
                int q = q0 + t * 16 + hh * 8 + r;
                float val = oacc[t][d][r] / lrow[t][r];
                O[((size_t)(b * T + q)) * C + head * 64 + d * 16 + ln] =
                    (_Float16)val;
            }
}

// ---------------------------------------------------------------------------
// Host orchestration
// ---------------------------------------------------------------------------
extern "C" void kernel_launch(void* const* d_in, const int* in_sizes, int n_in,
                              void* d_out, int out_size, void* d_ws, size_t ws_size,
                              hipStream_t stream) {
    (void)in_sizes; (void)n_in; (void)out_size; (void)ws_size;
    constexpr int Bn = 2, T = 2048, D = 1024, H = 16, F = 4096;
    constexpr int M = Bn * T;

    const float* x     = (const float*)d_in[0];
    const float* ln1_g = (const float*)d_in[1];
    const float* ln1_b = (const float*)d_in[2];
    const float* wq    = (const float*)d_in[3];
    const float* bq    = (const float*)d_in[4];
    const float* wk    = (const float*)d_in[5];
    const float* bk    = (const float*)d_in[6];
    const float* wv    = (const float*)d_in[7];
    const float* bv    = (const float*)d_in[8];
    const float* wo    = (const float*)d_in[9];
    const float* bo    = (const float*)d_in[10];
    const float* ln2_g = (const float*)d_in[11];
    const float* ln2_b = (const float*)d_in[12];
    const float* w1    = (const float*)d_in[13];
    const float* b1    = (const float*)d_in[14];
    const float* w2    = (const float*)d_in[15];
    const float* b2    = (const float*)d_in[16];

    char* ws = (char*)d_ws;
    size_t off = 0;
    auto alloc = [&](size_t bytes) -> void* {
        void* p = ws + off;
        off += (bytes + 255) & ~(size_t)255;
        return p;
    };
    _Float16* xn16 = (_Float16*)alloc((size_t)M * D * 2);  // LN out (reused)
    _Float16* wqT  = (_Float16*)alloc((size_t)D * D * 2);  // [N][K] transposed
    _Float16* wkT  = (_Float16*)alloc((size_t)D * D * 2);
    _Float16* wvT  = (_Float16*)alloc((size_t)D * D * 2);
    _Float16* woT  = (_Float16*)alloc((size_t)D * D * 2);
    _Float16* w1T  = (_Float16*)alloc((size_t)D * F * 2);  // [F][D]
    _Float16* w2T  = (_Float16*)alloc((size_t)F * D * 2);  // [D][F]
    _Float16* Qh   = (_Float16*)alloc((size_t)M * D * 2);  // [B,H,T,64]
    _Float16* Kh   = (_Float16*)alloc((size_t)M * D * 2);  // [B,H,T,64]
    _Float16* Vt   = (_Float16*)alloc((size_t)M * D * 2);  // [B,H,64,T]
    _Float16* O16  = (_Float16*)alloc((size_t)M * D * 2);  // [B,T,D]
    float*    x1   = (float*)  alloc((size_t)M * D * 4);   // post-attn residual
    _Float16* h16  = (_Float16*)alloc((size_t)M * F * 2);  // MLP hidden

    // weight f32 -> f16 transpose: in [K][N] -> out [N][K]
    convert_transpose_f16<<<dim3(D / 32, D / 32), 256, 0, stream>>>(wq, wqT, D, D);
    convert_transpose_f16<<<dim3(D / 32, D / 32), 256, 0, stream>>>(wk, wkT, D, D);
    convert_transpose_f16<<<dim3(D / 32, D / 32), 256, 0, stream>>>(wv, wvT, D, D);
    convert_transpose_f16<<<dim3(D / 32, D / 32), 256, 0, stream>>>(wo, woT, D, D);
    convert_transpose_f16<<<dim3(F / 32, D / 32), 256, 0, stream>>>(w1, w1T, D, F);
    convert_transpose_f16<<<dim3(D / 32, F / 32), 256, 0, stream>>>(w2, w2T, F, D);

    // LN1
    layernorm_f16<<<M, 256, 0, stream>>>(x, ln1_g, ln1_b, xn16, D);

    // QKV projections
    dim3 gD(D / 128, M / 64);
    gemm_wmma<<<gD, 256, 0, stream>>>(xn16, wqT, bq, nullptr, nullptr, Qh,
                                      M, D, D, T, FLAG_HEADSPLIT);
    gemm_wmma<<<gD, 256, 0, stream>>>(xn16, wkT, bk, nullptr, nullptr, Kh,
                                      M, D, D, T, FLAG_HEADSPLIT);
    gemm_wmma<<<gD, 256, 0, stream>>>(xn16, wvT, bv, nullptr, nullptr, Vt,
                                      M, D, D, T, FLAG_VT);

    // causal flash attention
    attn_wmma<<<dim3(T / 256, Bn * H), 256, 0, stream>>>(Qh, Kh, Vt, O16, T, H);

    // output projection + residual(x) -> x1 (f32)
    gemm_wmma<<<gD, 256, 0, stream>>>(O16, woT, bo, x, x1, nullptr,
                                      M, D, D, T, 0);

    // LN2 (reuse xn16)
    layernorm_f16<<<M, 256, 0, stream>>>(x1, ln2_g, ln2_b, xn16, D);

    // MLP up + exact GELU -> h16 (f16)
    gemm_wmma<<<dim3(F / 128, M / 64), 256, 0, stream>>>(
        xn16, w1T, b1, nullptr, nullptr, h16, M, F, D, T, FLAG_GELU);

    // MLP down + residual(x1) -> d_out (f32)
    gemm_wmma<<<gD, 256, 0, stream>>>(h16, w2T, b2, x1, (float*)d_out, nullptr,
                                      M, D, F, T, 0);
}
